// DeepseekV2MLAAttentionFusion_51462298140704
// MI455X (gfx1250) — matrix operand order repaired and checked by reading, not compiled, not run on hardware
//
#include <hip/hip_runtime.h>
#include <hip/hip_bf16.h>
#include <math.h>

// ---------------------------------------------------------------------------
// Problem constants (from reference)
// ---------------------------------------------------------------------------
#define T_     2048
#define HID_   4096
#define NH_    32
#define DN_    128
#define DR_    64
#define DV_    128
#define QLR_   1536
#define KVLR_  512
#define EPS_   1e-6f
#define THETA_ 10000.0f

typedef __attribute__((ext_vector_type(16))) __bf16 bf16x16;
typedef __attribute__((ext_vector_type(8)))  __bf16 bf16x8;
typedef __attribute__((ext_vector_type(8)))  float  f32x8;

static __device__ inline bf16x16 cat8(bf16x8 lo, bf16x8 hi) {
    return __builtin_shufflevector(lo, hi, 0,1,2,3,4,5,6,7,8,9,10,11,12,13,14,15);
}

static __device__ inline f32x8 wmma_bf16(bf16x16 a, bf16x16 b, f32x8 c) {
    // D = A(16x32 bf16) x B(32x16 bf16) + C(16x16 f32)
    return __builtin_amdgcn_wmma_f32_16x16x32_bf16(false, a, false, b, (short)0, c, false, false);
}

// Async global->LDS copy of one 16B chunk (per-lane), tracked by ASYNCcnt.
static __device__ inline void async_copy_b128(unsigned lds_addr, const void* gaddr) {
    asm volatile("global_load_async_to_lds_b128 %0, %1, off"
                 :: "v"(lds_addr), "v"(gaddr) : "memory");
}
static __device__ inline void wait_async0() {
    asm volatile("s_wait_asynccnt 0" ::: "memory");
}

// ---------------------------------------------------------------------------
// Generic GEMM: C[M,N] = A[M,K] @ B[K,N].  N/K/LDA compile-time so address
// math folds to immediates.  Block tile 128x64, 8 waves, each wave 32x32 via
// 2x2 bf16 WMMA accumulators, K stepped by 32.  All dims divide exactly.
// ---------------------------------------------------------------------------
#define GBM 128
#define GBN 64
#define GKS 32

template<int N, int K, int LDA>
__global__ __launch_bounds__(256) void gemm_bf16_kernel(
    const float* __restrict__ A, const float* __restrict__ B,
    float* __restrict__ C)
{
    __shared__ __align__(64) __bf16 Alds[GBM][GKS];   // [m][k]
    __shared__ __align__(64) __bf16 Blds[GBN][GKS];   // [n][k]  (transposed)

    const int tid  = threadIdx.x;
    const int lane = tid & 31;
    const int wave = tid >> 5;      // 0..7
    const int wm   = wave >> 1;     // 0..3  : 32-row sub-block
    const int wn   = wave & 1;      // 0..1  : 32-col sub-block
    const int half = lane >> 4;     // 0..1
    const int l16  = lane & 15;

    const int m0 = blockIdx.y * GBM;
    const int n0 = blockIdx.x * GBN;

    f32x8 acc[2][2] = {};

    for (int k0 = 0; k0 < K; k0 += GKS) {
        // ---- stage A tile (128x32) as bf16: 16 f32 per thread -------------
        {
            const int m  = tid >> 1;
            const int kh = (tid & 1) * 16;
            const float* src = A + (size_t)(m0 + m) * LDA + (k0 + kh);
            if (k0 + GKS < K) __builtin_prefetch(src + GKS, 0, 3);
            #pragma unroll
            for (int j = 0; j < 16; ++j) Alds[m][kh + j] = (__bf16)src[j];
        }
        // ---- stage B tile (32x64) transposed: 8 f32 per thread ------------
        {
            const int k  = tid >> 3;
            const int n8 = (tid & 7) * 8;
            const float* src = B + (size_t)(k0 + k) * N + (n0 + n8);
            if (k0 + GKS < K) __builtin_prefetch(src + (size_t)GKS * N, 0, 3);
            #pragma unroll
            for (int j = 0; j < 8; ++j) Blds[n8 + j][k] = (__bf16)src[j];
        }
        __syncthreads();

        #pragma unroll
        for (int mi = 0; mi < 2; ++mi) {
            // A fragment: lane holds row M=l16, 16 contiguous K at half*16
            const __bf16* arow = &Alds[wm * 32 + mi * 16 + l16][0];
            bf16x8 alo = *(const bf16x8*)(arow + half * 16);
            bf16x8 ahi = *(const bf16x8*)(arow + half * 16 + 8);
            bf16x16 afrag = cat8(alo, ahi);
            #pragma unroll
            for (int ni = 0; ni < 2; ++ni) {
                // B fragment: lane holds col N=l16, 16 contiguous K at half*16
                bf16x16 bfrag = *(const bf16x16*)(&Blds[wn * 32 + ni * 16 + l16][half * 16]);
                acc[mi][ni] = wmma_bf16(afrag, bfrag, acc[mi][ni]);
            }
        }
        __syncthreads();
    }

    // ---- epilogue: C/D layout -> row M = r + 8*half, col N = l16 ----------
    #pragma unroll
    for (int mi = 0; mi < 2; ++mi) {
        #pragma unroll
        for (int ni = 0; ni < 2; ++ni) {
            const int col = n0 + wn * 32 + ni * 16 + l16;
            float* cp = C + (size_t)(m0 + wm * 32 + mi * 16 + 8 * half) * N + col;
            #pragma unroll
            for (int r = 0; r < 8; ++r) cp[(size_t)r * N] = acc[mi][ni][r];
        }
    }
}

// ---------------------------------------------------------------------------
// RMSNorm over one row per block (in-place).  rowlen in {512,1536}, /256 exact.
// ---------------------------------------------------------------------------
__global__ __launch_bounds__(256) void rmsnorm_kernel(
    float* __restrict__ x, const float* __restrict__ g, int rowlen, int stride)
{
    __shared__ float red[256];
    float* p = x + (size_t)blockIdx.x * stride;
    const int tid = threadIdx.x;
    const int per = rowlen >> 8;        // 2 or 6
    float v[6];
    float ss = 0.f;
    for (int j = 0; j < per; ++j) { v[j] = p[tid + j * 256]; ss += v[j] * v[j]; }
    red[tid] = ss;
    __syncthreads();
    for (int s = 128; s > 0; s >>= 1) {
        if (tid < s) red[tid] += red[tid + s];
        __syncthreads();
    }
    const float rs = rsqrtf(red[0] / (float)rowlen + EPS_);
    for (int j = 0; j < per; ++j) p[tid + j * 256] = v[j] * rs * g[tid + j * 256];
}

// ---------------------------------------------------------------------------
// RoPE on k_pe (last 64 cols of kv rows), in place.  One thread per (t, pair).
// ---------------------------------------------------------------------------
__global__ __launch_bounds__(256) void kpe_rope_kernel(
    float* __restrict__ kvr, const long long* __restrict__ positions)
{
    const int id = blockIdx.x * 256 + threadIdx.x;   // T*32
    const int p  = id & 31;
    const int t  = id >> 5;
    const float pos = (float)positions[t];
    const float inv = powf(THETA_, -(float)(2 * p) / (float)DR_);
    const float a = pos * inv;
    const float c = cosf(a), s = sinf(a);
    float* base = kvr + (size_t)t * (KVLR_ + DR_) + KVLR_;
    const float x1 = base[2 * p], x2 = base[2 * p + 1];
    base[2 * p]     = x1 * c - x2 * s;
    base[2 * p + 1] = x2 * c + x1 * s;
}

// ---------------------------------------------------------------------------
// Q pack: rope q_pe and emit qf as bf16 in head-major layout [NH][T][192].
// ---------------------------------------------------------------------------
__global__ __launch_bounds__(256) void qpack_kernel(
    const float* __restrict__ qraw, const long long* __restrict__ positions,
    __bf16* __restrict__ qf)
{
    const int id  = blockIdx.x * 256 + threadIdx.x;  // T*NH*96
    const int p   = id % 96;
    const int rem = id / 96;
    const int h   = rem & (NH_ - 1);
    const int t   = rem >> 5;
    const float* src = qraw + (size_t)t * (NH_ * (DN_ + DR_)) + h * (DN_ + DR_);
    __bf16* dst = qf + (((size_t)h * T_) + t) * (DN_ + DR_);
    if (p < 64) {
        dst[2 * p]     = (__bf16)src[2 * p];
        dst[2 * p + 1] = (__bf16)src[2 * p + 1];
    } else {
        const int pp = p - 64;
        const float pos = (float)positions[t];
        const float inv = powf(THETA_, -(float)(2 * pp) / (float)DR_);
        const float a = pos * inv;
        const float c = cosf(a), s = sinf(a);
        const float x1 = src[DN_ + 2 * pp], x2 = src[DN_ + 2 * pp + 1];
        dst[DN_ + 2 * pp]     = (__bf16)(x1 * c - x2 * s);
        dst[DN_ + 2 * pp + 1] = (__bf16)(x2 * c + x1 * s);
    }
}

// ---------------------------------------------------------------------------
// K pack: kf[h][t][192] = {k_nope | k_pe(broadcast over heads)} as bf16.
// ---------------------------------------------------------------------------
__global__ __launch_bounds__(256) void kpack_kernel(
    const float* __restrict__ kvup, const float* __restrict__ kvr,
    __bf16* __restrict__ kf)
{
    const int id  = blockIdx.x * 256 + threadIdx.x;  // T*NH*192
    const int c   = id % 192;
    const int rem = id / 192;
    const int h   = rem & (NH_ - 1);
    const int t   = rem >> 5;
    const float val = (c < DN_)
        ? kvup[(size_t)t * (NH_ * (DN_ + DV_)) + h * (DN_ + DV_) + c]
        : kvr [(size_t)t * (KVLR_ + DR_) + KVLR_ + (c - DN_)];
    kf[(((size_t)h * T_) + t) * (DN_ + DR_) + c] = (__bf16)val;
}

// ---------------------------------------------------------------------------
// V transpose: vbT[h][dv][t] = bf16(kv_up[t][h*256 + 128 + dv]).
// LDS-tiled 32x32 so both global sides stay coalesced.  Grid (T/32, 4*NH).
// ---------------------------------------------------------------------------
__global__ __launch_bounds__(256) void vtrans_kernel(
    const float* __restrict__ kvup, __bf16* __restrict__ vbT)
{
    __shared__ __bf16 tile[32][33];
    const int t0  = blockIdx.x * 32;
    const int dv0 = (blockIdx.y & 3) * 32;
    const int h   = blockIdx.y >> 2;
    const int tx  = threadIdx.x & 31;
    const int ty4 = threadIdx.x >> 5;
    #pragma unroll
    for (int r = 0; r < 4; ++r) {
        const int ty = ty4 * 4 + r;   // token row
        tile[ty][tx] = (__bf16)kvup[(size_t)(t0 + ty) * (NH_ * (DN_ + DV_))
                                    + h * (DN_ + DV_) + DN_ + dv0 + tx];
    }
    __syncthreads();
    #pragma unroll
    for (int r = 0; r < 4; ++r) {
        const int ty = ty4 * 4 + r;   // dv row
        vbT[(((size_t)h * DV_) + dv0 + ty) * T_ + t0 + tx] = tile[tx][ty];
    }
}

// ---------------------------------------------------------------------------
// Flash attention, causal.  Grid (T/128, NH), 256 threads = 8 waves.
// Each wave owns 16 query rows; key tiles of 32; K/V tiles double-buffered in
// LDS via ASYNCcnt-tracked global_load_async_to_lds_b128; online softmax f32.
// ---------------------------------------------------------------------------
__global__ __launch_bounds__(256) void mla_attn_kernel(
    const __bf16* __restrict__ qf,   // [NH][T][192]
    const __bf16* __restrict__ kf,   // [NH][T][192]
    const __bf16* __restrict__ vbT,  // [NH][DV][T]   (pre-transposed)
    float* __restrict__ attn)        // [T][NH*128]
{
    __shared__ __align__(64) __bf16 Klds[2][32][192];   // [buf][key][d]     12KB*2
    __shared__ __align__(64) __bf16 Vlds[2][128][32];   // [buf][dv][key]     8KB*2
    __shared__ __align__(64) __bf16 Plds[8][16][32];    // per-wave P relayout

    const int tid  = threadIdx.x;
    const int lane = tid & 31;
    const int wave = tid >> 5;
    const int half = lane >> 4;
    const int l16  = lane & 15;
    const int h    = blockIdx.y;
    const int q0   = blockIdx.x * 128;
    const int qrowA = q0 + wave * 16 + l16;           // A-fragment row (M = l16)

    const float scale = 0.07216878364870323f;         // 1/sqrt(192)

    const unsigned kldsbase = (unsigned)(size_t)(&Klds[0][0][0]);
    const unsigned vldsbase = (unsigned)(size_t)(&Vlds[0][0][0]);

    // ---- async-stage one 32-key tile (K: 768 x 16B, V: 512 x 16B) ---------
    auto issue_tile = [&](int kt) {
        const int buf   = kt & 1;
        const int kbase = kt * 32;
        const char* ksrc = (const char*)(kf + (((size_t)h * T_) + kbase) * (DN_ + DR_));
        const unsigned kdst = kldsbase + (unsigned)buf * (32 * 192 * 2);
        #pragma unroll
        for (int c = 0; c < 3; ++c) {
            const int j = tid + c * 256;
            async_copy_b128(kdst + (unsigned)(j * 16), ksrc + (size_t)j * 16);
        }
        const unsigned vdst = vldsbase + (unsigned)buf * (128 * 32 * 2);
        #pragma unroll
        for (int c = 0; c < 2; ++c) {
            const int j  = tid + c * 256;
            const int dv = j >> 2, part = j & 3;
            const char* vsrc = (const char*)(vbT + (((size_t)h * DV_ + dv) * T_)
                                             + kbase + part * 8);
            async_copy_b128(vdst + (unsigned)((dv * 32 + part * 8) * 2), vsrc);
        }
    };

    // ---- load Q fragments once (6 x K=32 windows over d=192) --------------
    bf16x16 qfrag[6];
    const __bf16* qbase = qf + (((size_t)h * T_) + qrowA) * (DN_ + DR_);
    #pragma unroll
    for (int f = 0; f < 6; ++f) {
        bf16x8 lo = *(const bf16x8*)(qbase + f * 32 + half * 16);
        bf16x8 hi = *(const bf16x8*)(qbase + f * 32 + half * 16 + 8);
        qfrag[f] = cat8(lo, hi);
    }

    f32x8 oacc[8] = {};
    float mrow[8], lrow[8];
    #pragma unroll
    for (int r = 0; r < 8; ++r) { mrow[r] = -1e30f; lrow[r] = 0.f; }

    const int nkt = (q0 + 128) / 32;                  // causal key-tile bound
    issue_tile(0);
    for (int kt = 0; kt < nkt; ++kt) {
        const int buf   = kt & 1;
        const int kbase = kt * 32;
        wait_async0();          // own slice of tile kt landed
        __syncthreads();        // everyone's slice landed; prev reads done
        if (kt + 1 < nkt) issue_tile(kt + 1);   // overlap DMA with compute

        // ---- S = Q K^T for this wave's 16 rows x 32 keys ------------------
        f32x8 s0 = {}, s1 = {};
        #pragma unroll
        for (int f = 0; f < 6; ++f) {
            bf16x16 b0 = *(const bf16x16*)(&Klds[buf][l16][f * 32 + half * 16]);
            bf16x16 b1 = *(const bf16x16*)(&Klds[buf][16 + l16][f * 32 + half * 16]);
            s0 = wmma_bf16(qfrag[f], b0, s0);
            s1 = wmma_bf16(qfrag[f], b1, s1);
        }

        // ---- online softmax (per C-layout row r + 8*half) -----------------
        #pragma unroll
        for (int r = 0; r < 8; ++r) {
            const int q    = q0 + wave * 16 + r + 8 * half;
            const bool mk0 = (kbase + l16)      > q;
            const bool mk1 = (kbase + 16 + l16) > q;
            const float v0 = mk0 ? -1e30f : s0[r] * scale;
            const float v1 = mk1 ? -1e30f : s1[r] * scale;
            float tmax = fmaxf(v0, v1);
            #pragma unroll
            for (int off = 1; off < 16; off <<= 1)
                tmax = fmaxf(tmax, __shfl_xor(tmax, off, 32));
            const float mnew  = fmaxf(mrow[r], tmax);
            const float alpha = __expf(mrow[r] - mnew);
            const float p0 = mk0 ? 0.f : __expf(v0 - mnew);
            const float p1 = mk1 ? 0.f : __expf(v1 - mnew);
            float ps = p0 + p1;
            #pragma unroll
            for (int off = 1; off < 16; off <<= 1)
                ps += __shfl_xor(ps, off, 32);
            lrow[r] = lrow[r] * alpha + ps;
            mrow[r] = mnew;
            #pragma unroll
            for (int n = 0; n < 8; ++n) oacc[n][r] *= alpha;
            const int prow = r + 8 * half;
            Plds[wave][prow][l16]      = (__bf16)p0;
            Plds[wave][prow][16 + l16] = (__bf16)p1;
        }

        // ---- O += P V  (same-wave LDS is in-order; no barrier needed) -----
        {
            const __bf16* prow = &Plds[wave][l16][0];
            bf16x8 lo = *(const bf16x8*)(prow + half * 16);
            bf16x8 hi = *(const bf16x8*)(prow + half * 16 + 8);
            bf16x16 pfrag = cat8(lo, hi);
            #pragma unroll
            for (int n = 0; n < 8; ++n) {
                bf16x16 vfrag = *(const bf16x16*)(&Vlds[buf][n * 16 + l16][half * 16]);
                oacc[n] = wmma_bf16(pfrag, vfrag, oacc[n]);
            }
        }
        // next iteration's top barrier orders these reads vs. tile kt+2 DMA
    }

    // ---- normalize and write attn[T][NH*128] ------------------------------
    #pragma unroll
    for (int n = 0; n < 8; ++n) {
        const int col = h * DV_ + n * 16 + l16;
        #pragma unroll
        for (int r = 0; r < 8; ++r) {
            const int q = q0 + wave * 16 + r + 8 * half;
            attn[(size_t)q * (NH_ * DV_) + col] = oacc[n][r] / lrow[r];
        }
    }
}

// ---------------------------------------------------------------------------
// Host-side orchestration
// ---------------------------------------------------------------------------
extern "C" void kernel_launch(void* const* d_in, const int* in_sizes, int n_in,
                              void* d_out, int out_size, void* d_ws, size_t ws_size,
                              hipStream_t stream)
{
    const long long* positions = (const long long*)d_in[0];
    const float* hidden = (const float*)d_in[1];
    const float* w_qa   = (const float*)d_in[2];
    const float* w_kva  = (const float*)d_in[3];
    const float* g_qa   = (const float*)d_in[4];
    const float* w_qb   = (const float*)d_in[5];
    const float* g_kva  = (const float*)d_in[6];
    const float* w_kvb  = (const float*)d_in[7];
    const float* w_o    = (const float*)d_in[8];
    float* out = (float*)d_out;

    char* ws = (char*)d_ws;
    // Workspace layout (bytes); attn overlays the dead qa/kvr/qraw region.
    const size_t o_qa   = 0;                                   // 2048*1536*4
    const size_t o_kvr  = o_qa   + (size_t)T_ * QLR_ * 4;      // 2048*576*4
    const size_t o_qraw = o_kvr  + (size_t)T_ * (KVLR_ + DR_) * 4;
    const size_t o_kvup = o_qraw + (size_t)T_ * NH_ * (DN_ + DR_) * 4;
    const size_t o_qf   = o_kvup + (size_t)T_ * NH_ * (DN_ + DV_) * 4;
    const size_t o_kf   = o_qf   + (size_t)NH_ * T_ * (DN_ + DR_) * 2;
    const size_t o_vb   = o_kf   + (size_t)NH_ * T_ * (DN_ + DR_) * 2;
    const size_t o_attn = 0;                                   // reuse front region

    float*  qa   = (float*) (ws + o_qa);
    float*  kvr  = (float*) (ws + o_kvr);
    float*  qraw = (float*) (ws + o_qraw);
    float*  kvup = (float*) (ws + o_kvup);
    __bf16* qfb  = (__bf16*)(ws + o_qf);
    __bf16* kfb  = (__bf16*)(ws + o_kf);
    __bf16* vbT  = (__bf16*)(ws + o_vb);
    float*  attn = (float*) (ws + o_attn);

    // 1) q_a = hidden @ w_qa              [2048,1536]
    gemm_bf16_kernel<QLR_, HID_, HID_>
        <<<dim3(QLR_ / GBN, T_ / GBM), 256, 0, stream>>>(hidden, w_qa, qa);
    // 2) kv = hidden @ w_kva              [2048,576]
    gemm_bf16_kernel<KVLR_ + DR_, HID_, HID_>
        <<<dim3((KVLR_ + DR_) / GBN, T_ / GBM), 256, 0, stream>>>(hidden, w_kva, kvr);
    // 3) q_c = rmsnorm(q_a, g_qa)  (in place)
    rmsnorm_kernel<<<T_, 256, 0, stream>>>(qa, g_qa, QLR_, QLR_);
    // 4) kv_norm = rmsnorm(kv[:, :512], g_kva)  (in place)
    rmsnorm_kernel<<<T_, 256, 0, stream>>>(kvr, g_kva, KVLR_, KVLR_ + DR_);
    // 5) rope(k_pe)  (in place on kv[:, 512:576])
    kpe_rope_kernel<<<(T_ * 32) / 256, 256, 0, stream>>>(kvr, positions);
    // 6) q = q_c @ w_qb                   [2048,6144]
    gemm_bf16_kernel<NH_ * (DN_ + DR_), QLR_, QLR_>
        <<<dim3((NH_ * (DN_ + DR_)) / GBN, T_ / GBM), 256, 0, stream>>>(qa, w_qb, qraw);
    // 7) rope(q_pe) + pack qf -> bf16 [NH][T][192]
    qpack_kernel<<<((size_t)T_ * NH_ * 96) / 256, 256, 0, stream>>>(qraw, positions, qfb);
    // 8) kv_up = kv_norm @ w_kvb          [2048,8192]
    gemm_bf16_kernel<NH_ * (DN_ + DV_), KVLR_, KVLR_ + DR_>
        <<<dim3((NH_ * (DN_ + DV_)) / GBN, T_ / GBM), 256, 0, stream>>>(kvr, w_kvb, kvup);
    // 9) pack kf -> bf16 [NH][T][192]; transpose v -> bf16 [NH][DV][T]
    kpack_kernel<<<((size_t)T_ * NH_ * 192) / 256, 256, 0, stream>>>(kvup, kvr, kfb);
    vtrans_kernel<<<dim3(T_ / 32, 4 * NH_), 256, 0, stream>>>(kvup, vbT);
    // 10) causal flash attention -> attn  [2048,4096]
    mla_attn_kernel<<<dim3(T_ / 128, NH_), 256, 0, stream>>>(qfb, kfb, vbT, attn);
    // 11) out = attn @ w_o                [2048,4096]
    gemm_bf16_kernel<HID_, NH_ * DV_, NH_ * DV_>
        <<<dim3(HID_ / GBN, T_ / GBM), 256, 0, stream>>>(attn, w_o, out);
}
